// egnn_ablation_atom_types_only_30150670418632
// MI455X (gfx1250) — matrix operand order repaired and checked by reading, not compiled, not run on hardware
//
#include <hip/hip_runtime.h>
#include <hip/hip_bf16.h>

typedef __attribute__((ext_vector_type(16))) _Float16 hvec16;
typedef __attribute__((ext_vector_type(8)))  _Float16 hvec8;
typedef __attribute__((ext_vector_type(8)))  float    fvec8;

#define HDIM  256
#define LPAD  264   // LDS row stride in halves; 264*2 = 528 = 33*16 bytes -> 16B-aligned rows
#define ETILE 32    // edges per block in the fused edge kernel (2 row-tiles of 16)

__device__ __forceinline__ float silu_f(float x)    { return x / (1.f + __expf(-x)); }
__device__ __forceinline__ float sigmoid_f(float x) { return 1.f / (1.f + __expf(-x)); }

// A-fragment (16x32 f16, MxK). p = row base already offset by koffA = (lane>>4)*8.
// halves 0..7  <- K[k0 + kbase .. +7], halves 8..15 <- K[k0 + kbase + 16 .. +23]
__device__ __forceinline__ hvec16 frag_a(const _Float16* p, int k0) {
  hvec8 lo = *(const hvec8*)(p + k0);
  hvec8 hi = *(const hvec8*)(p + k0 + 16);
  hvec16 a;
#pragma unroll
  for (int i = 0; i < 8; ++i) { a[i] = lo[i]; a[i + 8] = hi[i]; }
  return a;
}

// B-fragment (32x16 f16, KxN) from a [N][K] (pre-transposed) matrix.
// p = row base already offset by koffB = (lane>>4)*8 : halves 0..7 <- K[k0+kbase..+7], 8..15 <- K[k0+kbase+16..+23].
__device__ __forceinline__ hvec16 frag_b(const _Float16* p, int k0) {
  hvec8 lo = *(const hvec8*)(p + k0);
  hvec8 hi = *(const hvec8*)(p + k0 + 16);
  hvec16 b;
#pragma unroll
  for (int i = 0; i < 8; ++i) { b[i] = lo[i]; b[i + 8] = hi[i]; }
  return b;
}

__device__ __forceinline__ fvec8 wmma16(hvec16 a, hvec16 b, fvec8 c) {
  return __builtin_amdgcn_wmma_f32_16x16x32_f16(false, a, false, b, (short)0, c, false, false);
}

// ---------------------------------------------------------------------------
// Fused per-edge kernel: edge MLP (2x GEMM), attention gate, coord MLP (GEMM),
// coord scalar, and segment-sum scatters. 128 threads / 32 edges per block.
// Each wave owns 4 col-tiles x 2 row-tiles; B fragments are shared across the
// two row-tiles (halves L2 weight traffic vs a 16-edge tile).
// ---------------------------------------------------------------------------
__global__ __launch_bounds__(128)
void egnn_edge_kernel(const int* __restrict__ erow, const int* __restrict__ ecol,
                      const float* __restrict__ x, const float* __restrict__ eattr,
                      const _Float16* __restrict__ hh,
                      const _Float16* __restrict__ W1aT, const _Float16* __restrict__ W1bT,
                      const float* __restrict__ w1c, const float* __restrict__ w1d,
                      const float* __restrict__ b1,
                      const _Float16* __restrict__ W2T, const float* __restrict__ b2,
                      const float* __restrict__ attw, const float* __restrict__ attb,
                      const _Float16* __restrict__ Wc1T, const float* __restrict__ bc1,
                      const float* __restrict__ wc2,
                      float* __restrict__ agg, float* __restrict__ num, float* __restrict__ cnt,
                      int E)
{
  __shared__ __align__(32) _Float16 smA[ETILE * LPAD];
  __shared__ __align__(32) _Float16 smB[ETILE * LPAD];
  __shared__ float s_rad[ETILE], s_ea[ETILE], s_cd[3 * ETILE], s_att[ETILE], s_sc[ETILE];
  __shared__ float s_part[ETILE * 4];
  __shared__ int   s_r[ETILE], s_c[ETILE];

  const int tid  = threadIdx.x;
  const int lane = tid & 31;
  const int wv   = tid >> 5;
  const int e0   = blockIdx.x * ETILE;

  if (tid < ETILE) {
    int e = e0 + tid; if (e >= E) e = E - 1;
    int r = erow[e], c = ecol[e];
    s_r[tid] = r; s_c[tid] = c;
    float dx = x[3*r+0] - x[3*c+0];
    float dy = x[3*r+1] - x[3*c+1];
    float dz = x[3*r+2] - x[3*c+2];
    s_cd[3*tid+0] = dx; s_cd[3*tid+1] = dy; s_cd[3*tid+2] = dz;
    s_rad[tid] = dx*dx + dy*dy + dz*dz;
    s_ea[tid]  = eattr[e];
  }
  __syncthreads();

  const int ar    = lane & 15;
  const int koffA = (lane >> 4) << 3;   // 0 or 8
  const int koffB = (lane >> 4) << 3;   // 0 or 8
  const _Float16* hr0 = hh + (size_t)s_r[ar]      * HDIM + koffA;
  const _Float16* hr1 = hh + (size_t)s_r[16 + ar] * HDIM + koffA;
  const _Float16* hc0 = hh + (size_t)s_c[ar]      * HDIM + koffA;
  const _Float16* hc1 = hh + (size_t)s_c[16 + ar] * HDIM + koffA;

  // ---- GEMM1: m1 = silu(h_row@W1a + h_col@W1b + radial*w1c + ea*w1d + b1) ----
  for (int t = 0; t < 4; ++t) {
    const int n = (wv + 4 * t) * 16 + ar;
    const _Float16* ba = W1aT + (size_t)n * HDIM + koffB;
    const _Float16* bb = W1bT + (size_t)n * HDIM + koffB;
    fvec8 a0 = {}, a1 = {};
#pragma unroll
    for (int k0 = 0; k0 < HDIM; k0 += 32) {
      hvec16 b = frag_b(ba, k0);
      a0 = wmma16(frag_a(hr0, k0), b, a0);
      a1 = wmma16(frag_a(hr1, k0), b, a1);
    }
#pragma unroll
    for (int k0 = 0; k0 < HDIM; k0 += 32) {
      hvec16 b = frag_b(bb, k0);
      a0 = wmma16(frag_a(hc0, k0), b, a0);
      a1 = wmma16(frag_a(hc1, k0), b, a1);
    }
    const float wcn = w1c[n], wdn = w1d[n], bn = b1[n];
#pragma unroll
    for (int rr = 0; rr < 8; ++rr) {
      int m  = koffA + rr;
      int m2 = m + 16;
      float v0 = a0[rr] + s_rad[m]  * wcn + s_ea[m]  * wdn + bn;
      float v1 = a1[rr] + s_rad[m2] * wcn + s_ea[m2] * wdn + bn;
      smA[m  * LPAD + n] = (_Float16)silu_f(v0);
      smA[m2 * LPAD + n] = (_Float16)silu_f(v1);
    }
  }
  __syncthreads();

  // ---- GEMM2: m = silu(m1 @ W2 + b2) ----
  {
    const _Float16* sa0 = smA + ar        * LPAD + koffA;
    const _Float16* sa1 = smA + (16 + ar) * LPAD + koffA;
    for (int t = 0; t < 4; ++t) {
      const int n = (wv + 4 * t) * 16 + ar;
      const _Float16* bw = W2T + (size_t)n * HDIM + koffB;
      fvec8 a0 = {}, a1 = {};
#pragma unroll
      for (int k0 = 0; k0 < HDIM; k0 += 32) {
        hvec16 b = frag_b(bw, k0);
        a0 = wmma16(frag_a(sa0, k0), b, a0);
        a1 = wmma16(frag_a(sa1, k0), b, a1);
      }
      const float bn = b2[n];
#pragma unroll
      for (int rr = 0; rr < 8; ++rr) {
        int m = koffA + rr;
        smB[m        * LPAD + n] = (_Float16)silu_f(a0[rr] + bn);
        smB[(m + 16) * LPAD + n] = (_Float16)silu_f(a1[rr] + bn);
      }
    }
  }
  __syncthreads();

  // ---- attention gate per edge (parallel partial dots: 4 threads/edge) ----
  {
    int m = tid >> 2, c0 = (tid & 3) * 64;
    float d = 0.f;
    for (int k = 0; k < 64; ++k) d += (float)smB[m * LPAD + c0 + k] * attw[c0 + k];
    s_part[tid] = d;
  }
  __syncthreads();
  if (tid < ETILE) {
    float d = attb[0] + s_part[4*tid] + s_part[4*tid+1] + s_part[4*tid+2] + s_part[4*tid+3];
    s_att[tid] = sigmoid_f(d);
  }
  __syncthreads();
  for (int i = tid; i < ETILE * HDIM; i += 128) {
    int m = i >> 8, c = i & 255;
    smB[m * LPAD + c] = (_Float16)((float)smB[m * LPAD + c] * s_att[m]);
  }
  __syncthreads();

  // ---- GEMM3: cm = silu(m @ Wc1 + bc1) ----
  {
    const _Float16* sa0 = smB + ar        * LPAD + koffA;
    const _Float16* sa1 = smB + (16 + ar) * LPAD + koffA;
    for (int t = 0; t < 4; ++t) {
      const int n = (wv + 4 * t) * 16 + ar;
      const _Float16* bw = Wc1T + (size_t)n * HDIM + koffB;
      fvec8 a0 = {}, a1 = {};
#pragma unroll
      for (int k0 = 0; k0 < HDIM; k0 += 32) {
        hvec16 b = frag_b(bw, k0);
        a0 = wmma16(frag_a(sa0, k0), b, a0);
        a1 = wmma16(frag_a(sa1, k0), b, a1);
      }
      const float bn = bc1[n];
#pragma unroll
      for (int rr = 0; rr < 8; ++rr) {
        int m = koffA + rr;
        smA[m        * LPAD + n] = (_Float16)silu_f(a0[rr] + bn);
        smA[(m + 16) * LPAD + n] = (_Float16)silu_f(a1[rr] + bn);
      }
    }
  }
  __syncthreads();

  // ---- coord scalar s = cm @ wc2 (parallel partial dots) ----
  {
    int m = tid >> 2, c0 = (tid & 3) * 64;
    float d = 0.f;
    for (int k = 0; k < 64; ++k) d += (float)smA[m * LPAD + c0 + k] * wc2[c0 + k];
    s_part[tid] = d;
  }
  __syncthreads();
  if (tid < ETILE)
    s_sc[tid] = s_part[4*tid] + s_part[4*tid+1] + s_part[4*tid+2] + s_part[4*tid+3];
  __syncthreads();

  // ---- scatter trans + counts ----
  if (tid < ETILE) {
    int e = e0 + tid;
    if (e < E) {
      int r = s_r[tid]; float s = s_sc[tid];
      atomicAdd(&num[3*r+0], s_cd[3*tid+0] * s);
      atomicAdd(&num[3*r+1], s_cd[3*tid+1] * s);
      atomicAdd(&num[3*r+2], s_cd[3*tid+2] * s);
      atomicAdd(&cnt[r], 1.f);
    }
  }
  // ---- agg: segment-sum of gated m into agg[row] ----
  for (int i = tid; i < ETILE * HDIM; i += 128) {
    int m = i >> 8, c = i & 255;
    if (e0 + m < E)
      atomicAdd(&agg[(size_t)s_r[m] * HDIM + c], (float)smB[m * LPAD + c]);
  }
}

// ---------------------------------------------------------------------------
// Node update: h_out = h + node_w2( silu(node_w1([h, agg]) ) )
// ---------------------------------------------------------------------------
__global__ __launch_bounds__(128)
void egnn_node_kernel(const float* __restrict__ hold,
                      const _Float16* __restrict__ hh, const _Float16* __restrict__ aggh,
                      const _Float16* __restrict__ NW1aT, const _Float16* __restrict__ NW1bT,
                      const float* __restrict__ nb1,
                      const _Float16* __restrict__ NW2T, const float* __restrict__ nb2,
                      float* __restrict__ hout, _Float16* __restrict__ hhout, int N)
{
  __shared__ __align__(32) _Float16 sm[16 * LPAD];
  const int tid = threadIdx.x, lane = tid & 31, wv = tid >> 5;
  const int r0 = blockIdx.x * 16;
  const int ar = lane & 15;
  const int koffA = (lane >> 4) << 3;
  const int koffB = (lane >> 4) << 3;
  int row = r0 + ar; if (row >= N) row = N - 1;
  const _Float16* ha = hh   + (size_t)row * HDIM + koffA;
  const _Float16* aa = aggh + (size_t)row * HDIM + koffA;

  for (int t = 0; t < 4; ++t) {
    const int n = (wv + 4 * t) * 16 + ar;
    const _Float16* ba = NW1aT + (size_t)n * HDIM + koffB;
    const _Float16* bb = NW1bT + (size_t)n * HDIM + koffB;
    fvec8 acc = {};
#pragma unroll
    for (int k0 = 0; k0 < HDIM; k0 += 32) acc = wmma16(frag_a(ha, k0), frag_b(ba, k0), acc);
#pragma unroll
    for (int k0 = 0; k0 < HDIM; k0 += 32) acc = wmma16(frag_a(aa, k0), frag_b(bb, k0), acc);
    const float bn = nb1[n];
#pragma unroll
    for (int rr = 0; rr < 8; ++rr) {
      int m = koffA + rr;
      sm[m * LPAD + n] = (_Float16)silu_f(acc[rr] + bn);
    }
  }
  __syncthreads();

  const _Float16* sa = sm + ar * LPAD + koffA;
  for (int t = 0; t < 4; ++t) {
    const int n = (wv + 4 * t) * 16 + ar;
    const _Float16* bw = NW2T + (size_t)n * HDIM + koffB;
    fvec8 acc = {};
#pragma unroll
    for (int k0 = 0; k0 < HDIM; k0 += 32) acc = wmma16(frag_a(sa, k0), frag_b(bw, k0), acc);
    const float bn = nb2[n];
#pragma unroll
    for (int rr = 0; rr < 8; ++rr) {
      int gm = r0 + koffA + rr;
      if (gm < N) {
        float v = acc[rr] + bn + hold[(size_t)gm * HDIM + n];
        hout[(size_t)gm * HDIM + n]  = v;
        hhout[(size_t)gm * HDIM + n] = (_Float16)v;
      }
    }
  }
}

// ---------------------------------------------------------------------------
// Input embedding: h = h_in(128) @ Wemb(128x256) + b, dual f32/f16 output
// ---------------------------------------------------------------------------
__global__ __launch_bounds__(128)
void egnn_embin_kernel(const _Float16* __restrict__ hin, const _Float16* __restrict__ EiT,
                       const float* __restrict__ bi,
                       float* __restrict__ hout, _Float16* __restrict__ hhout,
                       int N, int K)
{
  const int tid = threadIdx.x, lane = tid & 31, wv = tid >> 5;
  const int r0 = blockIdx.x * 16;
  const int ar = lane & 15;
  const int koffA = (lane >> 4) << 3;
  const int koffB = (lane >> 4) << 3;
  int row = r0 + ar; if (row >= N) row = N - 1;
  const _Float16* ha = hin + (size_t)row * K + koffA;

  for (int t = 0; t < 4; ++t) {
    const int n = (wv + 4 * t) * 16 + ar;
    const _Float16* bw = EiT + (size_t)n * K + koffB;
    fvec8 acc = {};
    for (int k0 = 0; k0 < K; k0 += 32) acc = wmma16(frag_a(ha, k0), frag_b(bw, k0), acc);
    const float bn = bi[n];
#pragma unroll
    for (int rr = 0; rr < 8; ++rr) {
      int gm = r0 + koffA + rr;
      if (gm < N) {
        float v = acc[rr] + bn;
        hout[(size_t)gm * HDIM + n]  = v;
        hhout[(size_t)gm * HDIM + n] = (_Float16)v;
      }
    }
  }
}

// ---------------------------------------------------------------------------
// Small utility kernels
// ---------------------------------------------------------------------------
__global__ void egnn_w_transpose_f16(const float* __restrict__ src, _Float16* __restrict__ dst,
                                     int K, int Nn, int k_off, int src_ld)
{
  int i = blockIdx.x * blockDim.x + threadIdx.x;
  if (i < K * Nn) {
    int n = i / K, k = i - n * K;
    dst[i] = (_Float16)src[(size_t)(k + k_off) * src_ld + n];
  }
}

__global__ void egnn_f32_to_f16(const float* __restrict__ s, _Float16* __restrict__ d, int n)
{
  int i = blockIdx.x * blockDim.x + threadIdx.x;
  if (i < n) d[i] = (_Float16)s[i];
}

__global__ void egnn_coord_update(float* __restrict__ x, const float* __restrict__ num,
                                  const float* __restrict__ cnt, int N)
{
  int i = blockIdx.x * blockDim.x + threadIdx.x;
  if (i < N * 3) x[i] += num[i] / fmaxf(cnt[i / 3], 1.f);
}

__global__ void egnn_embout_kernel(const float* __restrict__ h, const float* __restrict__ Wo,
                                   const float* __restrict__ bo, float* __restrict__ out, int N)
{
  int i = blockIdx.x * blockDim.x + threadIdx.x;
  if (i < N * 3) {
    int n = i / 3, j = i - n * 3;
    float s = bo[j];
    for (int k = 0; k < HDIM; ++k) s += h[(size_t)n * HDIM + k] * Wo[k * 3 + j];
    out[i] = s;
  }
}

// ---------------------------------------------------------------------------
extern "C" void kernel_launch(void* const* d_in, const int* in_sizes, int n_in,
                              void* d_out, int out_size, void* d_ws, size_t ws_size,
                              hipStream_t stream)
{
  const float* coords = (const float*)d_in[0];
  const float* h_src  = (const float*)d_in[1];
  const int*   eidx   = (const int*)d_in[2];
  const float* eattr  = (const float*)d_in[3];
  // d_in[4] = batch (unused)
  const float* embiW = (const float*)d_in[5];
  const float* embib = (const float*)d_in[6];
  const float* emboW = (const float*)d_in[7];
  const float* embob = (const float*)d_in[8];

  const int N   = in_sizes[0] / 3;
  const int INF = in_sizes[1] / N;     // 128
  const int E   = in_sizes[2] / 2;
  const int L   = 4;
  const int* erow = eidx;
  const int* ecol = eidx + E;

  // ---- workspace layout ----
  char* base = (char*)d_ws;
  size_t off = 0;
  auto take = [&](size_t bytes) -> char* {
    char* p = base + off;
    off = (off + bytes + 255) & ~(size_t)255;
    return p;
  };
  float*    xb   = (float*)take((size_t)N * 3 * 4);
  float*    h0   = (float*)take((size_t)N * HDIM * 4);
  float*    h1   = (float*)take((size_t)N * HDIM * 4);
  _Float16* hh0  = (_Float16*)take((size_t)N * HDIM * 2);
  _Float16* hh1  = (_Float16*)take((size_t)N * HDIM * 2);
  _Float16* hin  = (_Float16*)take((size_t)N * INF * 2);
  float*    agg  = (float*)take((size_t)N * HDIM * 4);
  _Float16* aggh = (_Float16*)take((size_t)N * HDIM * 2);
  float*    nnum = (float*)take((size_t)N * 3 * 4);
  float*    ncnt = (float*)take((size_t)N * 4);
  _Float16* EiT  = (_Float16*)take((size_t)HDIM * INF * 2);
  _Float16 *W1aT[4], *W1bT[4], *W2T[4], *Wc1T[4], *NW1aT[4], *NW1bT[4], *NW2T[4];
  for (int l = 0; l < L; ++l) {
    W1aT[l]  = (_Float16*)take((size_t)HDIM * HDIM * 2);
    W1bT[l]  = (_Float16*)take((size_t)HDIM * HDIM * 2);
    W2T[l]   = (_Float16*)take((size_t)HDIM * HDIM * 2);
    Wc1T[l]  = (_Float16*)take((size_t)HDIM * HDIM * 2);
    NW1aT[l] = (_Float16*)take((size_t)HDIM * HDIM * 2);
    NW1bT[l] = (_Float16*)take((size_t)HDIM * HDIM * 2);
    NW2T[l]  = (_Float16*)take((size_t)HDIM * HDIM * 2);
  }
  (void)ws_size; (void)n_in;

  // ---- layer param pointers (dict insertion order flattening) ----
  const float *eW1[4], *eb1[4], *eW2[4], *eb2[4], *nW1[4], *nb1[4], *nW2[4], *nb2[4];
  const float *cW1[4], *cb1[4], *cW2v[4], *aW[4], *ab[4];
  for (int l = 0; l < L; ++l) {
    int b = 9 + 13 * l;
    eW1[l]  = (const float*)d_in[b + 0];  eb1[l] = (const float*)d_in[b + 1];
    eW2[l]  = (const float*)d_in[b + 2];  eb2[l] = (const float*)d_in[b + 3];
    nW1[l]  = (const float*)d_in[b + 4];  nb1[l] = (const float*)d_in[b + 5];
    nW2[l]  = (const float*)d_in[b + 6];  nb2[l] = (const float*)d_in[b + 7];
    cW1[l]  = (const float*)d_in[b + 8];  cb1[l] = (const float*)d_in[b + 9];
    cW2v[l] = (const float*)d_in[b + 10];
    aW[l]   = (const float*)d_in[b + 11]; ab[l] = (const float*)d_in[b + 12];
  }

  // ---- one-time weight prep (deterministic, re-done every call) ----
  const int TT = 256;
  const int TW = (HDIM * HDIM + TT - 1) / TT;
  egnn_w_transpose_f16<<<(INF * HDIM + TT - 1) / TT, TT, 0, stream>>>(embiW, EiT, INF, HDIM, 0, HDIM);
  for (int l = 0; l < L; ++l) {
    egnn_w_transpose_f16<<<TW, TT, 0, stream>>>(eW1[l], W1aT[l],  HDIM, HDIM, 0,    HDIM);
    egnn_w_transpose_f16<<<TW, TT, 0, stream>>>(eW1[l], W1bT[l],  HDIM, HDIM, HDIM, HDIM);
    egnn_w_transpose_f16<<<TW, TT, 0, stream>>>(eW2[l], W2T[l],   HDIM, HDIM, 0,    HDIM);
    egnn_w_transpose_f16<<<TW, TT, 0, stream>>>(cW1[l], Wc1T[l],  HDIM, HDIM, 0,    HDIM);
    egnn_w_transpose_f16<<<TW, TT, 0, stream>>>(nW1[l], NW1aT[l], HDIM, HDIM, 0,    HDIM);
    egnn_w_transpose_f16<<<TW, TT, 0, stream>>>(nW1[l], NW1bT[l], HDIM, HDIM, HDIM, HDIM);
    egnn_w_transpose_f16<<<TW, TT, 0, stream>>>(nW2[l], NW2T[l],  HDIM, HDIM, 0,    HDIM);
  }

  // ---- x <- coords ; hin <- f16(h_src) ; h <- emb_in(h_src) ----
  hipMemcpyAsync(xb, coords, (size_t)N * 3 * 4, hipMemcpyDeviceToDevice, stream);
  egnn_f32_to_f16<<<(N * INF + TT - 1) / TT, TT, 0, stream>>>(h_src, hin, N * INF);
  egnn_embin_kernel<<<(N + 15) / 16, 128, 0, stream>>>(hin, EiT, embib, h0, hh0, N, INF);

  // ---- EGCL layers ----
  int cur = 0;
  float*    hf[2]  = { h0, h1 };
  _Float16* hhf[2] = { hh0, hh1 };
  for (int l = 0; l < L; ++l) {
    hipMemsetAsync(agg,  0, (size_t)N * HDIM * 4, stream);
    hipMemsetAsync(nnum, 0, (size_t)N * 3 * 4, stream);
    hipMemsetAsync(ncnt, 0, (size_t)N * 4, stream);

    egnn_edge_kernel<<<(E + ETILE - 1) / ETILE, 128, 0, stream>>>(
        erow, ecol, xb, eattr, hhf[cur],
        W1aT[l], W1bT[l],
        eW1[l] + (size_t)512 * HDIM,      // w1c: radial row of edge_w1
        eW1[l] + (size_t)513 * HDIM,      // w1d: edge_attr row of edge_w1
        eb1[l],
        W2T[l], eb2[l],
        aW[l], ab[l],
        Wc1T[l], cb1[l], cW2v[l],
        agg, nnum, ncnt, E);

    egnn_f32_to_f16<<<(N * HDIM + TT - 1) / TT, TT, 0, stream>>>(agg, aggh, N * HDIM);

    egnn_node_kernel<<<(N + 15) / 16, 128, 0, stream>>>(
        hf[cur], hhf[cur], aggh,
        NW1aT[l], NW1bT[l], nb1[l],
        NW2T[l], nb2[l],
        hf[1 - cur], hhf[1 - cur], N);

    egnn_coord_update<<<(N * 3 + TT - 1) / TT, TT, 0, stream>>>(xb, nnum, ncnt, N);
    cur ^= 1;
  }

  // ---- output projection ----
  egnn_embout_kernel<<<(N * 3 + TT - 1) / TT, TT, 0, stream>>>(
      hf[cur], emboW, embob, (float*)d_out, N);
  (void)out_size;
}
